// FNOBlock2d_29712583754400
// MI455X (gfx1250) — compile-verified, matrix operand order checked by reading, not run on hardware
//
#include <hip/hip_runtime.h>
#include <cmath>

// ---------------------------------------------------------------------------
// FNO block for gfx1250 (MI455X): partial-DFT-as-GEMM + per-mode complex MLP,
// all matmuls via v_wmma_f32_16x16x32_f16 (f32 accumulate).
// Twiddle fragments precomputed once into workspace (L2-resident), staged to
// LDS with GLOBAL_LOAD_ASYNC_TO_LDS (ASYNCcnt / s_wait_asynccnt).
// ---------------------------------------------------------------------------

#define BATCH 8
#define CIN   32
#define COUT  32
#define HDIM  512
#define WDIM  512
#define MODES 16
#define HID   128

typedef __attribute__((ext_vector_type(16))) _Float16 v16h;
typedef __attribute__((ext_vector_type(8)))  float    v8f;

union AFrag { v16h v; _Float16 e[16]; };
union CFrag { v8f  v; float     f[8]; };

#if defined(__has_builtin)
#if __has_builtin(__builtin_amdgcn_global_load_async_to_lds_b128) && \
    __has_builtin(__builtin_amdgcn_s_wait_asynccnt)
#define HAVE_ASYNC_LDS 1
#endif
#endif

#ifdef HAVE_ASYNC_LDS
// Matches the builtin's expected parameter type exactly:
//   __attribute__((__vector_size__(4 * sizeof(int)))) int *
typedef int vint4 __attribute__((vector_size(16)));
#endif

__device__ __forceinline__ int kmap16(int g, int j) {
  // 16-bit A/B 16x32 fragment K mapping (lanes 0-15: g=0, lanes 16-31: g=1)
  return (j < 8) ? (g * 8 + j) : (16 + g * 8 + (j - 8));
}

__device__ __forceinline__ v8f wmma16(v16h a, v16h b, v8f c) {
  return __builtin_amdgcn_wmma_f32_16x16x32_f16(false, a, false, b, (short)0, c,
                                                false, false);
}

#define PI2_OVER_N (6.283185307179586f / 512.0f)

// Workspace layout (bytes)
#define OFF_X1    ((size_t)0)                    // 16 MB  (131072 x 16 cplx)
#define OFF_XFT   ((size_t)16 * 1024 * 1024)     //  1 MB
#define OFF_OFT   ((size_t)17 * 1024 * 1024)     //  1 MB
#define OFF_Y1    ((size_t)18 * 1024 * 1024)     // 16 MB
#define OFF_LUT1  ((size_t)34 * 1024 * 1024)     // 32 KB fwd-W twiddle frags
#define OFF_BSPEC (OFF_LUT1 + (size_t)32768)     // 32 KB inv-W twiddle frags

// ---------------------------------------------------------------------------
// Kernel 0: build twiddle fragment LUTs (pre-swizzled f16 WMMA fragments).
//  lut1  : [plane(re/im)][kk(16)][lane(32)][j(16)] f16   (forward W DFT)
//  bspec : [nt(32)][lane(32)][j(16)] f16                 (inverse W DFT,
//          rows 0..15 = cos, 16..31 = -sin; alpha folded into A elsewhere)
// ---------------------------------------------------------------------------
__global__ void __launch_bounds__(256) k_init_lut(_Float16* __restrict__ lut1,
                                                  _Float16* __restrict__ bspec) {
  int id = blockIdx.x * 256 + threadIdx.x;  // 0..32767
  if (id < 16384) {
    int p = id >> 13;            // 0=re, 1=im
    int rem = id & 8191;
    int kk = rem >> 9;
    int lane = (rem >> 4) & 31;
    int j = rem & 15;
    int g = lane >> 4, n = lane & 15;
    int w = kk * 32 + kmap16(g, j);
    float t = PI2_OVER_N * (float)((n * w) & 511);
    lut1[id] = (_Float16)(p == 0 ? __cosf(t) : -__sinf(t));
  } else {
    int id2 = id - 16384;
    int nt = id2 >> 9;
    int lane = (id2 >> 4) & 31;
    int j = id2 & 15;
    int g = lane >> 4, ln = lane & 15;
    int k = kmap16(g, j);
    int kx = k & 15;
    int w = nt * 16 + ln;
    float t = PI2_OVER_N * (float)((kx * w) & 511);
    bspec[id2] = (_Float16)((k < 16) ? __cosf(t) : -__sinf(t));
  }
}

// ---------------------------------------------------------------------------
// Kernel 1: W-direction partial DFT.
//   x  : (B,CIN,H,W) f32, viewed as rows = B*CIN*H = 131072, K = 512
//   x1 : rows x 16 modes, complex (float2)
// Twiddle fragments async-copied from workspace into LDS once per block.
// ---------------------------------------------------------------------------
__global__ void __launch_bounds__(256) k_dft_w(const float* __restrict__ x,
                                               const _Float16* __restrict__ lutg,
                                               float* __restrict__ x1) {
  __shared__ __align__(32) _Float16 lutS[16384];  // 32 KB

  int tid = threadIdx.x;
#ifdef HAVE_ASYNC_LDS
  for (int r = 0; r < 8; ++r) {
    size_t off = (size_t)(tid + r * 256) * 16;  // bytes
    __builtin_amdgcn_global_load_async_to_lds_b128(
        (vint4*)((const char*)lutg + off), (vint4*)((char*)lutS + off), 0, 0);
  }
  __builtin_amdgcn_s_wait_asynccnt(0);
#else
  for (int r = 0; r < 8; ++r) {
    int i = tid + r * 256;
    ((uint4*)lutS)[i] = ((const uint4*)lutg)[i];
  }
#endif
  __syncthreads();

  const AFrag* lutRe = (const AFrag*)&lutS[0];     // [kk*32 + lane]
  const AFrag* lutIm = (const AFrag*)&lutS[8192];  // [kk*32 + lane]

  int wave = tid >> 5, lane = tid & 31;
  int g = lane >> 4, ln = lane & 15;
  int tile = blockIdx.x * 8 + wave;  // 8192 row tiles
  size_t row0 = (size_t)tile * 16;

  const float4* rowp = (const float4*)(x + (row0 + (size_t)ln) * WDIM);
  v8f accRe = {}, accIm = {};

  for (int kk = 0; kk < 16; ++kk) {
    if (kk + 1 < 16) __builtin_prefetch(rowp + (kk + 1) * 8 + g * 2, 0, 3);
    float4 f0 = rowp[kk * 8 + g * 2 + 0];      // K = g*8 + 0..3
    float4 f1 = rowp[kk * 8 + g * 2 + 1];      // K = g*8 + 4..7
    float4 f2 = rowp[kk * 8 + 4 + g * 2 + 0];  // K = 16 + g*8 + 0..3
    float4 f3 = rowp[kk * 8 + 4 + g * 2 + 1];  // K = 16 + g*8 + 4..7
    AFrag A;
    A.e[0]  = (_Float16)f0.x; A.e[1]  = (_Float16)f0.y;
    A.e[2]  = (_Float16)f0.z; A.e[3]  = (_Float16)f0.w;
    A.e[4]  = (_Float16)f1.x; A.e[5]  = (_Float16)f1.y;
    A.e[6]  = (_Float16)f1.z; A.e[7]  = (_Float16)f1.w;
    A.e[8]  = (_Float16)f2.x; A.e[9]  = (_Float16)f2.y;
    A.e[10] = (_Float16)f2.z; A.e[11] = (_Float16)f2.w;
    A.e[12] = (_Float16)f3.x; A.e[13] = (_Float16)f3.y;
    A.e[14] = (_Float16)f3.z; A.e[15] = (_Float16)f3.w;

    AFrag Br = lutRe[kk * 32 + lane];
    AFrag Bi = lutIm[kk * 32 + lane];
    accRe = wmma16(A.v, Br.v, accRe);
    accIm = wmma16(A.v, Bi.v, accIm);
  }

  CFrag cr, ci; cr.v = accRe; ci.v = accIm;
  float2* x1c = (float2*)x1;
  int n = lane & 15;
  for (int e = 0; e < 8; ++e) {
    int m = e + 8 * (lane >> 4);
    float2 o; o.x = cr.f[e]; o.y = ci.f[e];
    x1c[(row0 + m) * 16 + n] = o;
  }
}

// ---------------------------------------------------------------------------
// Kernel 2: H-direction DFT onto the two 16-row corners.
//   x1  : (img=B*CIN, H=512, 16) complex
//   xft : mode-major ((t*256 + ky*16 + kx), cin, bat) complex
// Per image: complex GEMM (32 ky) x (K=512 h) x (16 kx); one wave per image.
// ---------------------------------------------------------------------------
__global__ void __launch_bounds__(256) k_dft_h(const float* __restrict__ x1,
                                               float* __restrict__ xft) {
  int tid = threadIdx.x, wave = tid >> 5, lane = tid & 31;
  int g = lane >> 4, ln = lane & 15;
  int img = blockIdx.x * 8 + wave;  // 0..255 = b*CIN + c
  int b = img >> 5, c = img & 31;

  const float2* xr = (const float2*)x1 + (size_t)img * HDIM * 16;
  v8f accRe[2] = {{}, {}}, accIm[2] = {{}, {}};

  for (int kk = 0; kk < 16; ++kk) {
    AFrag Br, Bi;  // B = x1 slab: rows K=h, cols n=kx
    for (int j = 0; j < 16; ++j) {
      int hh = kk * 32 + kmap16(g, j);
      float2 v = xr[(size_t)hh * 16 + ln];
      Br.e[j] = (_Float16)v.x;
      Bi.e[j] = (_Float16)v.y;
    }
    for (int t = 0; t < 2; ++t) {
      int ky = (t == 0) ? ln : (496 + ln);  // A row m = ln
      AFrag Ar, Ai, An;                     // e^{-i t}: (cos, -sin), An = +sin
      for (int j = 0; j < 16; ++j) {
        int hh = kk * 32 + kmap16(g, j);
        float th = PI2_OVER_N * (float)((ky * hh) & 511);
        float cc = __cosf(th), ss = __sinf(th);
        Ar.e[j] = (_Float16)cc;
        Ai.e[j] = (_Float16)(-ss);
        An.e[j] = (_Float16)ss;
      }
      // (Ar + i Ai)(Br + i Bi): Re = Ar*Br + An*Bi ; Im = Ar*Bi + Ai*Br
      accRe[t] = wmma16(Ar.v, Br.v, accRe[t]);
      accRe[t] = wmma16(An.v, Bi.v, accRe[t]);
      accIm[t] = wmma16(Ar.v, Bi.v, accIm[t]);
      accIm[t] = wmma16(Ai.v, Br.v, accIm[t]);
    }
  }

  float2* xo = (float2*)xft;
  for (int t = 0; t < 2; ++t) {
    CFrag cr, ci; cr.v = accRe[t]; ci.v = accIm[t];
    for (int e = 0; e < 8; ++e) {
      int m = e + 8 * (lane >> 4);  // ky in tile
      int n = lane & 15;            // kx
      size_t idx = (((size_t)(t * 256 + m * 16 + n)) * CIN + c) * BATCH + b;
      float2 o; o.x = cr.f[e]; o.y = ci.f[e];
      xo[idx] = o;
    }
  }
}

// ---------------------------------------------------------------------------
// Kernel 3: per-mode complex MLP (Cin->HID->HID->Cout, ReLU on re/im between).
// One workgroup (8 waves) per (corner,mode); wave w owns output tile n=w*16.
// Intermediates in LDS.
// ---------------------------------------------------------------------------
__global__ void __launch_bounds__(256) k_mode_mlp(
    const float* __restrict__ xft,
    const float* __restrict__ wa0, const float* __restrict__ wa1,
    const float* __restrict__ wa2, const float* __restrict__ wb0,
    const float* __restrict__ wb1, const float* __restrict__ wb2,
    float* __restrict__ oft) {
  __shared__ float2 h0[HID][BATCH];
  __shared__ float2 h1[HID][BATCH];

  int tid = threadIdx.x, wave = tid >> 5, lane = tid & 31;
  int g = lane >> 4, ln = lane & 15;
  int mc = blockIdx.x;            // t*256 + mode
  int t = mc >> 8, mode = mc & 255;

  const float2* w0 = (const float2*)(t ? wb0 : wa0) + (size_t)mode * CIN * HID;
  const float2* w1 = (const float2*)(t ? wb1 : wa1) + (size_t)mode * HID * HID;
  const float2* w2 = (const float2*)(t ? wb2 : wa2) + (size_t)mode * HID * COUT;
  const float2* xm = (const float2*)xft + (size_t)mc * CIN * BATCH;

  // ---- layer 0: A = block (8 x Cin), rows >= 8 zero-padded
  AFrag Ar, Ai, An;
  for (int j = 0; j < 16; ++j) {
    int k = kmap16(g, j);
    float re = 0.f, im = 0.f;
    if (ln < 8) { float2 v = xm[(size_t)k * BATCH + ln]; re = v.x; im = v.y; }
    Ar.e[j] = (_Float16)re; Ai.e[j] = (_Float16)im; An.e[j] = (_Float16)(-im);
  }
  {
    int nt = wave;
    AFrag Wr, Wi;
    for (int j = 0; j < 16; ++j) {
      int k = kmap16(g, j);
      float2 v = w0[(size_t)k * HID + nt * 16 + ln];
      Wr.e[j] = (_Float16)v.x; Wi.e[j] = (_Float16)v.y;
    }
    v8f aR = {}, aI = {};
    aR = wmma16(Ar.v, Wr.v, aR); aR = wmma16(An.v, Wi.v, aR);
    aI = wmma16(Ar.v, Wi.v, aI); aI = wmma16(Ai.v, Wr.v, aI);
    CFrag cR, cI; cR.v = aR; cI.v = aI;
    for (int e = 0; e < 8; ++e) {
      int m = e + 8 * (lane >> 4);
      if (m < 8) {
        int n = nt * 16 + (lane & 15);
        float2 o; o.x = fmaxf(cR.f[e], 0.f); o.y = fmaxf(cI.f[e], 0.f);
        h0[n][m] = o;
      }
    }
  }
  __syncthreads();

  // ---- layer 1: (8 x HID) @ (HID x HID)
  {
    int nt = wave;
    v8f aR = {}, aI = {};
    for (int kk = 0; kk < 4; ++kk) {
      AFrag Hr, Hi, Hn;
      for (int j = 0; j < 16; ++j) {
        int k = kk * 32 + kmap16(g, j);
        float re = 0.f, im = 0.f;
        if (ln < 8) { float2 v = h0[k][ln]; re = v.x; im = v.y; }
        Hr.e[j] = (_Float16)re; Hi.e[j] = (_Float16)im; Hn.e[j] = (_Float16)(-im);
      }
      AFrag Wr, Wi;
      for (int j = 0; j < 16; ++j) {
        int k = kk * 32 + kmap16(g, j);
        float2 v = w1[(size_t)k * HID + nt * 16 + ln];
        Wr.e[j] = (_Float16)v.x; Wi.e[j] = (_Float16)v.y;
      }
      aR = wmma16(Hr.v, Wr.v, aR); aR = wmma16(Hn.v, Wi.v, aR);
      aI = wmma16(Hr.v, Wi.v, aI); aI = wmma16(Hi.v, Wr.v, aI);
    }
    CFrag cR, cI; cR.v = aR; cI.v = aI;
    for (int e = 0; e < 8; ++e) {
      int m = e + 8 * (lane >> 4);
      if (m < 8) {
        int n = wave * 16 + (lane & 15);
        float2 o; o.x = fmaxf(cR.f[e], 0.f); o.y = fmaxf(cI.f[e], 0.f);
        h1[n][m] = o;
      }
    }
  }
  __syncthreads();

  // ---- layer 2: (8 x HID) @ (HID x Cout), waves 0..1 only, no activation
  if (wave < 2) {
    int nt = wave;
    v8f aR = {}, aI = {};
    for (int kk = 0; kk < 4; ++kk) {
      AFrag Hr, Hi, Hn;
      for (int j = 0; j < 16; ++j) {
        int k = kk * 32 + kmap16(g, j);
        float re = 0.f, im = 0.f;
        if (ln < 8) { float2 v = h1[k][ln]; re = v.x; im = v.y; }
        Hr.e[j] = (_Float16)re; Hi.e[j] = (_Float16)im; Hn.e[j] = (_Float16)(-im);
      }
      AFrag Wr, Wi;
      for (int j = 0; j < 16; ++j) {
        int k = kk * 32 + kmap16(g, j);
        float2 v = w2[(size_t)k * COUT + nt * 16 + ln];
        Wr.e[j] = (_Float16)v.x; Wi.e[j] = (_Float16)v.y;
      }
      aR = wmma16(Hr.v, Wr.v, aR); aR = wmma16(Hn.v, Wi.v, aR);
      aI = wmma16(Hr.v, Wi.v, aI); aI = wmma16(Hi.v, Wr.v, aI);
    }
    CFrag cR, cI; cR.v = aR; cI.v = aI;
    float2* om = (float2*)oft + (size_t)mc * COUT * BATCH;
    for (int e = 0; e < 8; ++e) {
      int m = e + 8 * (lane >> 4);
      if (m < 8) {
        int n = nt * 16 + (lane & 15);
        float2 o; o.x = cR.f[e]; o.y = cI.f[e];
        om[(size_t)n * BATCH + m] = o;
      }
    }
  }
}

// ---------------------------------------------------------------------------
// Kernel 4: inverse H DFT. y1[b,co,h,kx] = sum over 32 retained ky of
// oft * e^{+2*pi*i*ky*h/512}.  Per (b,co): (512 h) x (K=32) x (16 kx) complex.
// ---------------------------------------------------------------------------
__global__ void __launch_bounds__(256) k_idft_h(const float* __restrict__ oft,
                                                float* __restrict__ y1) {
  int tid = threadIdx.x, wave = tid >> 5, lane = tid & 31;
  int g = lane >> 4, ln = lane & 15;
  int img = blockIdx.x;  // b*COUT + co
  int b = img >> 5, co = img & 31;

  const float2* om = (const float2*)oft;
  AFrag Br, Bi;  // B = oft slab: rows K = ky-index (t,ky), cols n = kx
  for (int j = 0; j < 16; ++j) {
    int k = kmap16(g, j);
    int t = k >> 4, ky = k & 15;
    size_t idx = (((size_t)(t * 256 + ky * 16 + ln)) * COUT + co) * BATCH + b;
    float2 v = om[idx];
    Br.e[j] = (_Float16)v.x; Bi.e[j] = (_Float16)v.y;
  }

  float2* yo = (float2*)y1 + (size_t)img * HDIM * 16;
  for (int i = 0; i < 4; ++i) {
    int mtile = wave * 4 + i;
    int hrow = mtile * 16 + ln;  // A row m = ln
    AFrag Ar, Ai, An;            // e^{+i t}: (cos, +sin), An = -sin
    for (int j = 0; j < 16; ++j) {
      int k = kmap16(g, j);
      int kyA = (k < 16) ? k : (480 + k);  // 496..511 for second corner
      float th = PI2_OVER_N * (float)((kyA * hrow) & 511);
      float cc = __cosf(th), ss = __sinf(th);
      Ar.e[j] = (_Float16)cc; Ai.e[j] = (_Float16)ss; An.e[j] = (_Float16)(-ss);
    }
    v8f aR = {}, aI = {};
    aR = wmma16(Ar.v, Br.v, aR); aR = wmma16(An.v, Bi.v, aR);
    aI = wmma16(Ar.v, Bi.v, aI); aI = wmma16(Ai.v, Br.v, aI);
    CFrag cR, cI; cR.v = aR; cI.v = aI;
    for (int e = 0; e < 8; ++e) {
      int m = e + 8 * (lane >> 4);
      int n = lane & 15;
      float2 o; o.x = cR.f[e]; o.y = cI.f[e];
      yo[(size_t)(mtile * 16 + m) * 16 + n] = o;
    }
  }
}

// ---------------------------------------------------------------------------
// Kernel 5: fused inverse W DFT + 1x1 bypass + bias + GELU.
// Per (b,h): out[b,:,h,:] (32 x 512). Both GEMMs share M=cout, N=w, K=32 and
// one f32 accumulator. x slab async-staged to LDS; inverse-DFT B fragments
// loaded pre-swizzled from the L2-resident LUT; 1/(H*W) and Hermitian
// doubling folded into the spectral A fragment.
// ---------------------------------------------------------------------------
__global__ void __launch_bounds__(256) k_fuse_out(
    const float* __restrict__ x, const float* __restrict__ y1,
    const _Float16* __restrict__ bspec, const float* __restrict__ wbyp,
    const float* __restrict__ bbyp, float* __restrict__ out) {
  __shared__ __align__(16) float xs[CIN][WDIM];  // 64 KB

  int tid = threadIdx.x, wave = tid >> 5, lane = tid & 31;
  int g = lane >> 4, ln = lane & 15;
  int b = blockIdx.x >> 9, hh = blockIdx.x & 511;

  // stage x[b,:,hh,:] (32 x 512 f32) into LDS
#ifdef HAVE_ASYNC_LDS
  for (int r = 0; r < 16; ++r) {
    int idx = tid + r * 256;  // 0..4095 float4 chunks
    int ci = idx >> 7, c4 = idx & 127;
    const float* src =
        x + ((size_t)(b * CIN + ci) * HDIM + hh) * WDIM + (size_t)c4 * 4;
    __builtin_amdgcn_global_load_async_to_lds_b128(
        (vint4*)src, (vint4*)&xs[ci][c4 * 4], 0, 0);
  }
  __builtin_amdgcn_s_wait_asynccnt(0);
#else
  for (int r = 0; r < 16; ++r) {
    int idx = tid + r * 256;
    int ci = idx >> 7, c4 = idx & 127;
    const float4* src =
        (const float4*)(x + ((size_t)(b * CIN + ci) * HDIM + hh) * WDIM);
    *((float4*)&xs[ci][c4 * 4]) = src[c4];
  }
#endif
  __syncthreads();

  // A fragments (2 M-tiles of cout)
  AFrag Abyp[2], Aspec[2];
  const float2* yc = (const float2*)y1;
  for (int mt = 0; mt < 2; ++mt) {
    int co = mt * 16 + ln;
    // cache the 8 spectral coefficients this lane group needs (kx = g*8+i)
    float2 yv[8];
    for (int i = 0; i < 8; ++i)
      yv[i] = yc[((size_t)(b * COUT + co) * HDIM + hh) * 16 + (g * 8 + i)];
    for (int j = 0; j < 16; ++j) {
      int k = kmap16(g, j);
      Abyp[mt].e[j] = (_Float16)wbyp[co * CIN + k];
      int kx = k & 15;
      float alpha = ((kx == 0) ? 1.0f : 2.0f) * (1.0f / (512.0f * 512.0f));
      float val = (k < 16) ? yv[kx - g * 8].x : yv[kx - g * 8].y;
      Aspec[mt].e[j] = (_Float16)(alpha * val);
    }
  }

  const AFrag* bsG = (const AFrag*)bspec;  // [nt*32 + lane], pre-swizzled
  for (int i = 0; i < 4; ++i) {
    int nt = wave * 4 + i;  // 32 N-tiles of w
    int wcol = nt * 16 + ln;
    AFrag Bs = bsG[nt * 32 + lane];
    AFrag Bb;
    for (int j = 0; j < 16; ++j) {
      int k = kmap16(g, j);
      Bb.e[j] = (_Float16)xs[k][wcol];
    }
    for (int mt = 0; mt < 2; ++mt) {
      v8f acc = {};
      acc = wmma16(Abyp[mt].v, Bb.v, acc);   // bypass conv
      acc = wmma16(Aspec[mt].v, Bs.v, acc);  // spectral irfft along W
      CFrag cf; cf.v = acc;
      for (int e = 0; e < 8; ++e) {
        int m = e + 8 * (lane >> 4);
        int co = mt * 16 + m;
        int w = nt * 16 + (lane & 15);
        float v = cf.f[e] + bbyp[co];
        float t3 = v * v * v;
        float gl = 0.5f * v * (1.0f + tanhf(0.7978845608f * (v + 0.044715f * t3)));
        out[((size_t)(b * COUT + co) * HDIM + hh) * WDIM + w] = gl;
      }
    }
  }
}

// ---------------------------------------------------------------------------
extern "C" void kernel_launch(void* const* d_in, const int* in_sizes, int n_in,
                              void* d_out, int out_size, void* d_ws,
                              size_t ws_size, hipStream_t stream) {
  (void)in_sizes; (void)n_in; (void)out_size; (void)ws_size;
  const float* x    = (const float*)d_in[0];
  const float* wa0  = (const float*)d_in[1];
  const float* wa1  = (const float*)d_in[2];
  const float* wa2  = (const float*)d_in[3];
  const float* wb0  = (const float*)d_in[4];
  const float* wb1  = (const float*)d_in[5];
  const float* wb2  = (const float*)d_in[6];
  const float* wbyp = (const float*)d_in[7];
  const float* bbyp = (const float*)d_in[8];
  float* out = (float*)d_out;

  char* ws = (char*)d_ws;
  float* x1  = (float*)(ws + OFF_X1);
  float* xft = (float*)(ws + OFF_XFT);
  float* oft = (float*)(ws + OFF_OFT);
  float* y1  = (float*)(ws + OFF_Y1);
  _Float16* lut1  = (_Float16*)(ws + OFF_LUT1);
  _Float16* bspec = (_Float16*)(ws + OFF_BSPEC);

  k_init_lut<<< 128, 256, 0, stream>>>(lut1, bspec);
  k_dft_w   <<<1024, 256, 0, stream>>>(x, lut1, x1);
  k_dft_h   <<<  32, 256, 0, stream>>>(x1, xft);
  k_mode_mlp<<< 512, 256, 0, stream>>>(xft, wa0, wa1, wa2, wb0, wb1, wb2, oft);
  k_idft_h  <<< 256, 256, 0, stream>>>(oft, y1);
  k_fuse_out<<<4096, 256, 0, stream>>>(x, y1, bspec, wbyp, bbyp, out);
}
